// FrameTransformer_12034498363736
// MI455X (gfx1250) — matrix operand order repaired and checked by reading, not compiled, not run on hardware
//
#include <hip/hip_runtime.h>
#include <hip/hip_bf16.h>

// ---------------------------------------------------------------------------
// FrameTransformer forward for MI455X (gfx1250, wave32).
// - All contractions via v_wmma_f32_16x16x32_f16 (one wave = 16x64 strip).
// - B-tiles staged into LDS by the Tensor Data Mover (tensor_load_to_lds,
//   TENSORcnt sync) when available; cooperative-copy fallback otherwise.
// ---------------------------------------------------------------------------

typedef _Float16 half_t;
typedef __attribute__((ext_vector_type(16))) _Float16 v16h;
typedef __attribute__((ext_vector_type(8)))  float    v8f;
typedef __attribute__((ext_vector_type(4)))  unsigned int u32x4;
typedef __attribute__((ext_vector_type(4)))  int      i32x4;
typedef __attribute__((ext_vector_type(8)))  int      i32x8;

struct __align__(16) U4 { unsigned int a, b, c, d; };
union F16x16 { v16h v; U4 u[2]; };

#define CB    2
#define CCH   8
#define CBINS 1024
#define CW    1024
#define CFF   2048
#define CHD   256
#define CM    (CB * CW)   // 2048 fused rows

#if __has_builtin(__builtin_amdgcn_tensor_load_to_lds) && \
    __has_builtin(__builtin_amdgcn_s_wait_tensorcnt)
#define USE_TDM 1
#else
#define USE_TDM 0
#endif

#if USE_TDM
// Issue one TDM 2D tile load: tile_n rows x tile_k f16 elements from
// gptr (row stride = stride_elems f16) into LDS byte offset lds_off.
// D# layout per CDNA5 ISA ch.8 (group0: flags/lds/global/type, group1: dims).
__device__ __forceinline__ void tdm_load_tile(unsigned lds_off, const void* gptr,
                                              unsigned k_rem, unsigned n_rem,
                                              unsigned tile_k, unsigned tile_n,
                                              unsigned stride_elems)
{
    unsigned long long ga = (unsigned long long)(__UINTPTR_TYPE__)gptr;
    u32x4 g0;
    g0[0] = 1u;                                              // count=1, user D#
    g0[1] = lds_off;                                         // lds_addr (bytes)
    g0[2] = (unsigned)(ga & 0xFFFFFFFFu);                    // global_addr lo
    g0[3] = (unsigned)((ga >> 32) & 0x1FFFFFFu) | (2u << 30);// addr hi | type=2
    i32x8 g1;
    g1[0] = (int)(1u << 16);                                 // data_size=1 (2B)
    g1[1] = (int)((k_rem & 0xFFFFu) << 16);                  // tensor_dim0 lo16
    g1[2] = (int)(((k_rem >> 16) & 0xFFFFu) |
                  ((n_rem & 0xFFFFu) << 16));                // dim0 hi | dim1 lo
    g1[3] = (int)(((n_rem >> 16) & 0xFFFFu) |
                  ((tile_k & 0xFFFFu) << 16));               // dim1 hi | tile_dim0
    g1[4] = (int)(tile_n & 0xFFFFu);                         // tile_dim1, dim2=0
    g1[5] = (int)stride_elems;                               // dim0_stride lo32
    g1[6] = 0;                                               // stride hi / dim1_stride
    g1[7] = 0;
    i32x4 z4 = {0, 0, 0, 0};
#if defined(__clang_major__) && (__clang_major__ >= 23)
    i32x8 z8 = {0, 0, 0, 0, 0, 0, 0, 0};
    __builtin_amdgcn_tensor_load_to_lds(g0, g1, z4, z4, z8, 0);
#else
    __builtin_amdgcn_tensor_load_to_lds(g0, g1, z4, z4, 0);
#endif
}
#endif

// ---------------------------------------------------------------------------
// WMMA GEMM:  C[M,N] = epilogue( A[M,K](f16,row-major) * B ),
// B supplied as Wt[N][K] row-major (B[k][n] = Wt[n*ldb+k]).
// Block = 256 thr = 8 waves = 128 rows x 64 cols; B strip (64xK) is shared by
// the whole block and staged 64x64 at a time in LDS via TDM.
// Requires: M % 128 == 0, N % 64 == 0, K % 64 == 0 (true for every call here).
// ---------------------------------------------------------------------------
__global__ void __launch_bounds__(256)
k_gemm(const half_t* __restrict__ A, int lda,
       const half_t* __restrict__ Bw, int ldb,
       float* __restrict__ C, half_t* __restrict__ C16, int ldc,
       const float* __restrict__ bias,
       const float* __restrict__ res, int ldr,
       const float* __restrict__ addT, int ldt,
       float scale, int act, int M, int N, int K)
{
    const int lane = threadIdx.x & 31;
    const int wave = threadIdx.x >> 5;
    const int hi   = lane >> 4;          // lane half -> K subgroup
    const int l15  = lane & 15;
    const int m0   = blockIdx.y * 128 + wave * 16;
    const int n0   = blockIdx.x * 64;

    __shared__ half_t ldsB[64 * 64];     // 8 KB tile: [64 n-rows][64 k]

    v8f acc[4] = {};
    const half_t* arow = A + (size_t)(m0 + l15) * lda;

    for (int kb = 0; kb < K; kb += 64) {
        __syncthreads();                 // previous tile fully consumed
#if USE_TDM
        if (wave == 0) {
            tdm_load_tile((unsigned)(__UINTPTR_TYPE__)&ldsB[0],
                          Bw + (size_t)n0 * ldb + kb,
                          (unsigned)(K - kb), (unsigned)(N - n0),
                          64u, 64u, (unsigned)ldb);
            __builtin_amdgcn_s_wait_tensorcnt(0);
        }
#else
        {   // cooperative fallback: 256 thr x 16 halves = 64x64 tile
            const int row = threadIdx.x >> 2;         // 0..63
            const int col = (threadIdx.x & 3) * 16;   // 0,16,32,48
            const half_t* gp = Bw + (size_t)(n0 + row) * ldb + kb + col;
            *reinterpret_cast<U4*>(&ldsB[row * 64 + col])     =
                *reinterpret_cast<const U4*>(gp);
            *reinterpret_cast<U4*>(&ldsB[row * 64 + col + 8]) =
                *reinterpret_cast<const U4*>(gp + 8);
        }
#endif
        __syncthreads();

#pragma unroll
        for (int kk = 0; kk < 64; kk += 32) {
            // A fragment: lane l row = l&15; halves = two contiguous 16B runs.
            F16x16 fa;
            fa.u[0] = *reinterpret_cast<const U4*>(arow + kb + kk + 8 * hi);
            fa.u[1] = *reinterpret_cast<const U4*>(arow + kb + kk + 16 + 8 * hi);
            if (kk == 0 && kb + 64 < K)
                __builtin_prefetch(arow + kb + 64, 0, 0);  // global_prefetch_b8
#pragma unroll
            for (int t = 0; t < 4; ++t) {
                // B fragment from LDS: lane col n=16t+l15, halves = two 8-K runs (k=8hi.., 16+8hi..).
                const half_t* bp = &ldsB[(16 * t + l15) * 64 + kk + 8 * hi];
                F16x16 fb;
                fb.u[0] = *reinterpret_cast<const U4*>(bp);
                fb.u[1] = *reinterpret_cast<const U4*>(bp + 16);
                acc[t] = __builtin_amdgcn_wmma_f32_16x16x32_f16(
                    false, fa.v, false, fb.v, (short)0, acc[t], false, false);
            }
        }
    }

#pragma unroll
    for (int t = 0; t < 4; ++t) {
        const int n = n0 + 16 * t + l15;
#pragma unroll
        for (int r = 0; r < 8; ++r) {
            const int m = m0 + 8 * hi + r;
            float v = acc[t][r];
            if (addT) v += addT[(size_t)n * ldt + m];   // transposed rel-pos bias
            v *= scale;
            if (bias) v += bias[n];
            if (act == 1)      { v = fmaxf(v, 0.f); }
            else if (act == 2) { v = fmaxf(v, 0.f); v *= v; }
            if (res) v += res[(size_t)m * ldr + n];
            const size_t o = (size_t)m * ldc + n;
            if (C)   C[o]   = v;
            if (C16) C16[o] = (half_t)v;
        }
    }
}

// ---------------------------------------------------------------------------
// Elementwise / reduction helpers
// ---------------------------------------------------------------------------

__global__ void k_inproject(const float* __restrict__ x,
                            const float* __restrict__ win,
                            float* __restrict__ xf, int n)
{
    int i = blockIdx.x * blockDim.x + threadIdx.x;
    if (i >= n) return;
    int bw = i / CBINS, f = i % CBINS;
    int b = bw / CW, w = bw % CW;
    float s = 0.f;
#pragma unroll
    for (int c = 0; c < CCH; ++c)
        s += x[(((size_t)b * CCH + c) * CBINS + f) * CW + w] * win[c];
    xf[i] = s;
}

__global__ void __launch_bounds__(256)
k_layernorm(const float* __restrict__ in, const float* __restrict__ g,
            const float* __restrict__ b, float* __restrict__ outF,
            half_t* __restrict__ outH, int C)
{
    const int row = blockIdx.x;
    const float* xr = in + (size_t)row * C;
    __shared__ float red[256];
    const int tid = threadIdx.x;

    float s = 0.f;
    for (int i = tid; i < C; i += 256) s += xr[i];
    red[tid] = s; __syncthreads();
    for (int o = 128; o > 0; o >>= 1) { if (tid < o) red[tid] += red[tid + o]; __syncthreads(); }
    const float mean = red[0] / C;
    __syncthreads();

    s = 0.f;
    for (int i = tid; i < C; i += 256) { float d = xr[i] - mean; s += d * d; }
    red[tid] = s; __syncthreads();
    for (int o = 128; o > 0; o >>= 1) { if (tid < o) red[tid] += red[tid + o]; __syncthreads(); }
    const float inv = rsqrtf(red[0] / C + 1e-5f);

    for (int i = tid; i < C; i += 256) {
        float v = (xr[i] - mean) * inv * g[i] + b[i];
        size_t o = (size_t)row * C + i;
        if (outF) outF[o] = v;
        if (outH) outH[o] = (half_t)v;
    }
}

__global__ void k_glu_gate(float* __restrict__ xf, const float* __restrict__ g, int n)
{
    int i = blockIdx.x * blockDim.x + threadIdx.x;
    if (i >= n) return;
    int m = i / CBINS, f = i % CBINS;
    float ga = g[(size_t)m * (2 * CBINS) + f];
    float gb = g[(size_t)m * (2 * CBINS) + CBINS + f];
    xf[i] += ga * (1.f / (1.f + expf(-gb)));
}

__global__ void k_dwconv(const float* __restrict__ in, const float* __restrict__ wgt,
                         const float* __restrict__ bias, half_t* __restrict__ outH,
                         float* __restrict__ outF, int C, int taps, int n)
{
    int i = blockIdx.x * blockDim.x + threadIdx.x;
    if (i >= n) return;
    int m = i / C, f = i % C;
    int w = m % CW, rowbase = m - w;
    const int pad = taps >> 1;
    float s = bias ? bias[f] : 0.f;
    for (int t = 0; t < taps; ++t) {
        int w2 = w + t - pad;
        if (w2 >= 0 && w2 < CW)
            s += in[(size_t)(rowbase + w2) * C + f] * wgt[f * taps + t];
    }
    if (outH) outH[i] = (half_t)s;
    if (outF) outF[i] = s;
}

__global__ void k_build_shift3(const float* __restrict__ h, half_t* __restrict__ out, int n)
{
    int i = blockIdx.x * blockDim.x + threadIdx.x;
    if (i >= n) return;
    int m = i / (3 * CBINS), r = i % (3 * CBINS);
    int t = r / CBINS, c = r % CBINS;
    int w = m % CW, w2 = w + t - 1;
    float v = 0.f;
    if (w2 >= 0 && w2 < CW) v = h[(size_t)(m - w + w2) * CBINS + c];
    out[i] = (half_t)v;
}

__global__ void k_reorder_c1r(const float* __restrict__ w, half_t* __restrict__ out, int n)
{
    int i = blockIdx.x * blockDim.x + threadIdx.x;
    if (i >= n) return;
    int o = i / (3 * CBINS), r = i % (3 * CBINS);
    int t = r / CBINS, c = r % CBINS;
    out[i] = (half_t)w[((size_t)o * CBINS + c) * 3 + t];
}

__global__ void k_transpose_er(const float* __restrict__ er, half_t* __restrict__ out, int n)
{
    int i = blockIdx.x * blockDim.x + threadIdx.x;
    if (i >= n) return;
    int p = i / CHD, h = i % CHD;
    out[i] = (half_t)er[(size_t)h * CW + p];
}

// vT[z][nn][k] = v[(b*W + k)*BINS + band*HD + nn]  (z = b*4 + band)
__global__ void k_transpose_v(const half_t* __restrict__ v, half_t* __restrict__ vt, int n)
{
    int i = blockIdx.x * blockDim.x + threadIdx.x;
    if (i >= n) return;
    int z = i / (CHD * CW), r = i % (CHD * CW);
    int nn = r / CW, k = r % CW;
    int b = z >> 2, band = z & 3;
    vt[i] = v[((size_t)b * CW + k) * CBINS + band * CHD + nn];
}

__global__ void k_cvt_f16(const float* __restrict__ in, half_t* __restrict__ out, int n)
{
    int i = blockIdx.x * blockDim.x + threadIdx.x;
    if (i < n) out[i] = (half_t)in[i];
}

__global__ void __launch_bounds__(256)
k_softmax(const float* __restrict__ S, half_t* __restrict__ out)
{
    const int row = blockIdx.x;
    const float* xr = S + (size_t)row * CW;
    __shared__ float red[256];
    const int tid = threadIdx.x;

    float mx = -1e30f;
    for (int i = tid; i < CW; i += 256) mx = fmaxf(mx, xr[i]);
    red[tid] = mx; __syncthreads();
    for (int o = 128; o > 0; o >>= 1) { if (tid < o) red[tid] = fmaxf(red[tid], red[tid + o]); __syncthreads(); }
    mx = red[0]; __syncthreads();

    float s = 0.f;
    for (int i = tid; i < CW; i += 256) s += expf(xr[i] - mx);
    red[tid] = s; __syncthreads();
    for (int o = 128; o > 0; o >>= 1) { if (tid < o) red[tid] += red[tid + o]; __syncthreads(); }
    const float inv = 1.f / red[0];

    for (int i = tid; i < CW; i += 256)
        out[(size_t)row * CW + i] = (half_t)(expf(xr[i] - mx) * inv);
}

__global__ void k_out_transpose(const float* __restrict__ xf, float* __restrict__ out, int n)
{
    int i = blockIdx.x * blockDim.x + threadIdx.x;
    if (i >= n) return;
    int bw = i / CBINS, f = i % CBINS;
    int b = bw / CW, w = bw % CW;
    out[((size_t)b * CBINS + f) * CW + w] = xf[i];
}

// ---------------------------------------------------------------------------
// Host-side orchestration
// ---------------------------------------------------------------------------
static inline dim3 g1(int n) { return dim3((n + 255) / 256); }

static void gemm(hipStream_t s, const half_t* A, int lda, const half_t* Bw, int ldb,
                 float* C, half_t* C16, int ldc, const float* bias,
                 const float* res, int ldr, const float* addT, int ldt,
                 float scale, int act, int M, int N, int K)
{
    dim3 grid(N / 64, M / 128, 1);
    k_gemm<<<grid, 256, 0, s>>>(A, lda, Bw, ldb, C, C16, ldc,
                                bias, res, ldr, addT, ldt, scale, act, M, N, K);
}

extern "C" void kernel_launch(void* const* d_in, const int* in_sizes, int n_in,
                              void* d_out, int out_size, void* d_ws, size_t ws_size,
                              hipStream_t stream)
{
    const float* x      = (const float*)d_in[0];
    const float* w_in   = (const float*)d_in[1];
    const float* ln1_g  = (const float*)d_in[2];
    const float* ln1_b  = (const float*)d_in[3];
    const float* glu_w  = (const float*)d_in[4];
    const float* ln2_g  = (const float*)d_in[5];
    const float* ln2_b  = (const float*)d_in[6];
    const float* c1L_w  = (const float*)d_in[7];
    const float* c1R_w  = (const float*)d_in[8];
    const float* ln3_g  = (const float*)d_in[9];
    const float* ln3_b  = (const float*)d_in[10];
    const float* c1M_dw = (const float*)d_in[11];
    const float* c1M_pw = (const float*)d_in[12];
    const float* ln4_g  = (const float*)d_in[13];
    const float* ln4_b  = (const float*)d_in[14];
    const float* qw     = (const float*)d_in[15];
    const float* qb     = (const float*)d_in[16];
    const float* qcw    = (const float*)d_in[17];
    const float* qcb    = (const float*)d_in[18];
    const float* kw     = (const float*)d_in[19];
    const float* kb_    = (const float*)d_in[20];
    const float* kcw    = (const float*)d_in[21];
    const float* kcb    = (const float*)d_in[22];
    const float* vw     = (const float*)d_in[23];
    const float* vb     = (const float*)d_in[24];
    const float* vcw    = (const float*)d_in[25];
    const float* vcb    = (const float*)d_in[26];
    const float* ow     = (const float*)d_in[27];
    const float* ob     = (const float*)d_in[28];
    const float* er     = (const float*)d_in[29];
    const float* ln5_g  = (const float*)d_in[30];
    const float* ln5_b  = (const float*)d_in[31];
    const float* c2_w   = (const float*)d_in[32];
    const float* c3_w   = (const float*)d_in[33];

    unsigned char* wsp = (unsigned char*)d_ws;
    size_t off = 0;
    auto alloc = [&](size_t bytes) -> void* {
        off = (off + 255) & ~(size_t)255;
        void* r = wsp + off; off += bytes; return r;
    };
    const int M = CM;  // 2048
    float*  xf     = (float*)alloc((size_t)M * CBINS * 4);
    float*  gbuf   = (float*)alloc((size_t)M * CFF * 4);
    float*  tmpB   = (float*)alloc((size_t)M * CFF * 4);
    half_t* h16    = (half_t*)alloc((size_t)M * CFF * 2);
    half_t* ashift = (half_t*)alloc((size_t)M * 3 * CBINS * 2);
    half_t* t16    = (half_t*)alloc((size_t)M * CFF * 2);
    half_t* q16    = (half_t*)alloc((size_t)M * CBINS * 2);
    half_t* k16    = (half_t*)alloc((size_t)M * CBINS * 2);
    half_t* v16    = (half_t*)alloc((size_t)M * CBINS * 2);
    half_t* vT16   = (half_t*)alloc((size_t)8 * CHD * CW * 2);
    half_t* o16    = (half_t*)alloc((size_t)M * CBINS * 2);
    float*  Pbuf   = (float*)alloc((size_t)8 * CW * CW * 4);
    float*  Sbuf   = (float*)alloc((size_t)8 * CW * CW * 4);
    half_t* attn16 = (half_t*)alloc((size_t)8 * CW * CW * 2);
    half_t* glu16  = (half_t*)alloc((size_t)2 * CBINS * CBINS * 2);
    half_t* c1L16  = (half_t*)alloc((size_t)CFF * CBINS * 2);
    half_t* c1R16  = (half_t*)alloc((size_t)(CFF / 4) * 3 * CBINS * 2);
    half_t* pw16   = (half_t*)alloc((size_t)CBINS * CFF * 2);
    half_t* qw16   = (half_t*)alloc((size_t)CBINS * CBINS * 2);
    half_t* kw16   = (half_t*)alloc((size_t)CBINS * CBINS * 2);
    half_t* vw16   = (half_t*)alloc((size_t)CBINS * CBINS * 2);
    half_t* ow16   = (half_t*)alloc((size_t)CBINS * CBINS * 2);
    half_t* erT16  = (half_t*)alloc((size_t)CW * CHD * 2);
    half_t* c2w16  = (half_t*)alloc((size_t)CFF * CBINS * 2);
    half_t* c3w16  = (half_t*)alloc((size_t)CBINS * CFF * 2);

    // ---- weight conversion ----
    int n;
    n = 2 * CBINS * CBINS;     k_cvt_f16<<<g1(n), 256, 0, stream>>>(glu_w, glu16, n);
    n = CFF * CBINS;           k_cvt_f16<<<g1(n), 256, 0, stream>>>(c1L_w, c1L16, n);
    n = (CFF / 4) * 3 * CBINS; k_reorder_c1r<<<g1(n), 256, 0, stream>>>(c1R_w, c1R16, n);
    n = CBINS * CFF;           k_cvt_f16<<<g1(n), 256, 0, stream>>>(c1M_pw, pw16, n);
    n = CBINS * CBINS;         k_cvt_f16<<<g1(n), 256, 0, stream>>>(qw, qw16, n);
    n = CBINS * CBINS;         k_cvt_f16<<<g1(n), 256, 0, stream>>>(kw, kw16, n);
    n = CBINS * CBINS;         k_cvt_f16<<<g1(n), 256, 0, stream>>>(vw, vw16, n);
    n = CBINS * CBINS;         k_cvt_f16<<<g1(n), 256, 0, stream>>>(ow, ow16, n);
    n = CW * CHD;              k_transpose_er<<<g1(n), 256, 0, stream>>>(er, erT16, n);
    n = CFF * CBINS;           k_cvt_f16<<<g1(n), 256, 0, stream>>>(c2_w, c2w16, n);
    n = CBINS * CFF;           k_cvt_f16<<<g1(n), 256, 0, stream>>>(c3_w, c3w16, n);

    // ---- 1) in-project ----
    n = M * CBINS;
    k_inproject<<<g1(n), 256, 0, stream>>>(x, w_in, xf, n);

    // ---- 2) GLU block ----
    k_layernorm<<<M, 256, 0, stream>>>(xf, ln1_g, ln1_b, nullptr, h16, CBINS);
    gemm(stream, h16, CBINS, glu16, CBINS, gbuf, nullptr, 2 * CBINS,
         nullptr, nullptr, 0, nullptr, 0, 1.f, 0, M, 2 * CBINS, CBINS);
    n = M * CBINS;
    k_glu_gate<<<g1(n), 256, 0, stream>>>(xf, gbuf, n);

    // ---- 3) conv feed-forward block ----
    k_layernorm<<<M, 256, 0, stream>>>(xf, ln2_g, ln2_b, tmpB, h16, CBINS);
    gemm(stream, h16, CBINS, c1L16, CBINS, gbuf, nullptr, CFF,
         nullptr, nullptr, 0, nullptr, 0, 1.f, 1, M, CFF, CBINS);
    n = M * 3 * CBINS;
    k_build_shift3<<<g1(n), 256, 0, stream>>>(tmpB, ashift, n);
    gemm(stream, ashift, 3 * CBINS, c1R16, 3 * CBINS, gbuf, nullptr, CFF,
         nullptr, gbuf, CFF, nullptr, 0, 1.f, 1, M, CFF / 4, 3 * CBINS);
    k_layernorm<<<M, 256, 0, stream>>>(gbuf, ln3_g, ln3_b, tmpB, nullptr, CFF);
    n = M * CFF;
    k_dwconv<<<g1(n), 256, 0, stream>>>(tmpB, c1M_dw, nullptr, t16, nullptr, CFF, 9, n);
    gemm(stream, t16, CFF, pw16, CFF, xf, nullptr, CBINS,
         nullptr, xf, CBINS, nullptr, 0, 1.f, 0, M, CBINS, CFF);

    // ---- 4) multiband attention ----
    k_layernorm<<<M, 256, 0, stream>>>(xf, ln4_g, ln4_b, nullptr, h16, CBINS);
    n = M * CBINS;
    gemm(stream, h16, CBINS, qw16, CBINS, tmpB, nullptr, CBINS,
         qb, nullptr, 0, nullptr, 0, 1.f, 0, M, CBINS, CBINS);
    k_dwconv<<<g1(n), 256, 0, stream>>>(tmpB, qcw, qcb, q16, nullptr, CBINS, 3, n);
    gemm(stream, h16, CBINS, kw16, CBINS, tmpB, nullptr, CBINS,
         kb_, nullptr, 0, nullptr, 0, 1.f, 0, M, CBINS, CBINS);
    k_dwconv<<<g1(n), 256, 0, stream>>>(tmpB, kcw, kcb, k16, nullptr, CBINS, 3, n);
    gemm(stream, h16, CBINS, vw16, CBINS, tmpB, nullptr, CBINS,
         vb, nullptr, 0, nullptr, 0, 1.f, 0, M, CBINS, CBINS);
    k_dwconv<<<g1(n), 256, 0, stream>>>(tmpB, vcw, vcb, v16, nullptr, CBINS, 3, n);
    n = 8 * CHD * CW;
    k_transpose_v<<<g1(n), 256, 0, stream>>>(v16, vT16, n);

    const float inv_sqrt_c = 0.03125f;  // 1/sqrt(1024)
    for (int z = 0; z < 8; ++z) {
        const int b = z >> 2, band = z & 3;
        const size_t boff = (size_t)b * CW * CBINS + (size_t)band * CHD;
        // P[j,i] = sum_h Q[j,h] * er[h,i]
        gemm(stream, q16 + boff, CBINS, erT16, CHD,
             Pbuf + (size_t)z * CW * CW, nullptr, CW,
             nullptr, nullptr, 0, nullptr, 0, 1.f, 0, CW, CW, CHD);
        // S[qi,ki] = (Q.Kt + P[ki,qi]) / sqrt(c)
        gemm(stream, q16 + boff, CBINS, k16 + boff, CBINS,
             Sbuf + (size_t)z * CW * CW, nullptr, CW,
             nullptr, nullptr, 0, Pbuf + (size_t)z * CW * CW, CW,
             inv_sqrt_c, 0, CW, CW, CHD);
    }
    k_softmax<<<8 * CW, 256, 0, stream>>>(Sbuf, attn16);
    for (int z = 0; z < 8; ++z) {
        const int b = z >> 2, band = z & 3;
        const size_t boff = (size_t)b * CW * CBINS + (size_t)band * CHD;
        // O = A @ V  via pre-transposed V (Wt form [HD][W])
        gemm(stream, attn16 + (size_t)z * CW * CW, CW, vT16 + (size_t)z * CHD * CW, CW,
             nullptr, o16 + boff, CBINS,
             nullptr, nullptr, 0, nullptr, 0, 1.f, 0, CW, CHD, CW);
    }
    gemm(stream, o16, CBINS, ow16, CBINS, xf, nullptr, CBINS,
         ob, xf, CBINS, nullptr, 0, 1.f, 0, M, CBINS, CBINS);

    // ---- 5) squared-ReLU FFN ----
    k_layernorm<<<M, 256, 0, stream>>>(xf, ln5_g, ln5_b, nullptr, h16, CBINS);
    gemm(stream, h16, CBINS, c2w16, CBINS, nullptr, t16, CFF,
         nullptr, nullptr, 0, nullptr, 0, 1.f, 2, M, CFF, CBINS);
    gemm(stream, t16, CFF, c3w16, CFF, xf, nullptr, CBINS,
         nullptr, xf, CBINS, nullptr, 0, 1.f, 0, M, CBINS, CFF);

    // ---- 6) final transpose ----
    n = M * CBINS;
    k_out_transpose<<<g1(n), 256, 0, stream>>>(xf, (float*)d_out, n);
}
